// Attention_55027120996441
// MI455X (gfx1250) — matrix operand order repaired and checked by reading, not compiled, not run on hardware
//
#include <hip/hip_runtime.h>
#include <hip/hip_bf16.h>

#define B_ 32
#define S_ 2048
#define H_ 512

typedef __attribute__((ext_vector_type(16))) __bf16 v16bf;
typedef __attribute__((ext_vector_type(8)))  float  v8f;

union BFrag { v16bf v; unsigned short u[16]; };

__device__ __forceinline__ unsigned short f2bf(float f) {
    unsigned int u = __float_as_uint(f);
    u += 0x7FFFu + ((u >> 16) & 1u);          // round-to-nearest-even
    return (unsigned short)(u >> 16);
}

// ---------------------------------------------------------------------------
// Kernel 1: W_enc (fp32, [H,H] row-major, row o = weights of output feature o)
//           -> bf16, same layout. 512 KB result stays L2-resident.
// ---------------------------------------------------------------------------
__global__ void wenc_to_bf16_kernel(const float* __restrict__ W,
                                    unsigned short* __restrict__ Wbf) {
    int i = blockIdx.x * blockDim.x + threadIdx.x;     // 65536 threads, 4 elem each
    float4 f = ((const float4*)W)[i];
    unsigned int lo = (unsigned int)f2bf(f.x) | ((unsigned int)f2bf(f.y) << 16);
    unsigned int hi = (unsigned int)f2bf(f.z) | ((unsigned int)f2bf(f.w) << 16);
    ((uint2*)Wbf)[i] = make_uint2(lo, hi);
}

// ---------------------------------------------------------------------------
// Kernel 2: dfb[b][o] = s_t_hat[b] . W_dec[o] + b_dec[o] + b_enc[o]
// ---------------------------------------------------------------------------
__global__ void dec_kernel(const float* __restrict__ s_t_hat,
                           const float* __restrict__ W_dec,
                           const float* __restrict__ b_dec,
                           const float* __restrict__ b_enc,
                           float* __restrict__ dfb) {
    int b = blockIdx.x;
    int tid = threadIdx.x;                              // 256 threads
    __shared__ float sv[H_];
    sv[tid]       = s_t_hat[b * H_ + tid];
    sv[tid + 256] = s_t_hat[b * H_ + tid + 256];
    __syncthreads();
    #pragma unroll
    for (int oo = 0; oo < 2; ++oo) {
        int o = tid + oo * 256;
        const float* wr = W_dec + (size_t)o * H_;
        float acc = 0.f;
        #pragma unroll 8
        for (int h = 0; h < H_; ++h) acc = fmaf(sv[h], wr[h], acc);
        dfb[b * H_ + o] = acc + b_dec[o] + b_enc[o];
    }
}

// ---------------------------------------------------------------------------
// Kernel 3: scores[b,s] = W_v . tanh(W_enc @ enc[b,s] + dfb[b] + cov[b,s]*W_c)
// One wave = 16 tokens. A tile (16x512 bf16) preloaded into 16 WMMA fragments.
// 32 N-tiles x 16 K-steps of v_wmma_f32_16x16x32_bf16, fused tanh epilogue.
// ---------------------------------------------------------------------------
__global__ void __launch_bounds__(256, 1)
scores_kernel(const float* __restrict__ enc,
              const unsigned short* __restrict__ Wbf,
              const float* __restrict__ dfb,
              const float* __restrict__ coverage,
              const float* __restrict__ W_c,     // [H,1]
              const float* __restrict__ W_v,     // [1,H]
              float* __restrict__ scores) {
    const int wave = threadIdx.x >> 5;
    const int lane = threadIdx.x & 31;
    const int half = lane >> 4;                  // 0: lanes 0-15, 1: lanes 16-31
    const int b      = blockIdx.x >> 4;          // 16 blocks per batch row
    const int s_base = (blockIdx.x & 15) << 7;   // 128 tokens per block
    const int tok0   = s_base + wave * 16;
    const int trow   = tok0 + (lane & 15);       // A-matrix row owned by this lane

    // ---- Preload A: 16 fragments covering K = 0..511 for 16 tokens ----
    // A layout (contiguous K, matching B): lanes 0-15 hold K=kb..kb+15,
    // lanes 16-31 hold K=kb+16..kb+31.
    v16bf afrag[16];
    const float* rowp = enc + (size_t)(b * S_ + trow) * H_;
    #pragma unroll
    for (int kk = 0; kk < 16; ++kk) {
        int c0 = kk * 32 + half * 16;
        float4 a0 = *(const float4*)(rowp + c0);
        float4 a1 = *(const float4*)(rowp + c0 + 4);
        float4 a2 = *(const float4*)(rowp + c0 + 8);
        float4 a3 = *(const float4*)(rowp + c0 + 12);
        BFrag fa;
        fa.u[0]  = f2bf(a0.x); fa.u[1]  = f2bf(a0.y); fa.u[2]  = f2bf(a0.z); fa.u[3]  = f2bf(a0.w);
        fa.u[4]  = f2bf(a1.x); fa.u[5]  = f2bf(a1.y); fa.u[6]  = f2bf(a1.z); fa.u[7]  = f2bf(a1.w);
        fa.u[8]  = f2bf(a2.x); fa.u[9]  = f2bf(a2.y); fa.u[10] = f2bf(a2.z); fa.u[11] = f2bf(a2.w);
        fa.u[12] = f2bf(a3.x); fa.u[13] = f2bf(a3.y); fa.u[14] = f2bf(a3.z); fa.u[15] = f2bf(a3.w);
        afrag[kk] = fa.v;
    }

    // Coverage for the 8 tokens this lane accumulates (C/D layout: VGPR r,
    // lanes 0-15 -> M=r, lanes 16-31 -> M=r+8).
    float cov[8];
    #pragma unroll
    for (int r = 0; r < 8; ++r)
        cov[r] = coverage[(size_t)b * S_ + tok0 + 8 * half + r];

    float sc[8];
    #pragma unroll
    for (int r = 0; r < 8; ++r) sc[r] = 0.f;

    // ---- Main loop over 32 N-tiles of 16 output features ----
    #pragma unroll 1
    for (int n = 0; n < 32; ++n) {
        const int o = n * 16 + (lane & 15);
        // B layout (32x16 KxN bf16): lane n holds column o; lanes 0-15 carry
        // K=kb..kb+15, lanes 16-31 K=kb+16..31 -> one contiguous 32B read.
        const unsigned short* wrow = Wbf + (size_t)o * H_;
        if (n < 31)
            __builtin_prefetch(Wbf + (size_t)(o + 16) * H_, 0, 1);

        v8f acc = {};
        #pragma unroll
        for (int kk = 0; kk < 16; ++kk) {
            v16bf bfrag = *(const v16bf*)(wrow + kk * 32 + half * 16);
            acc = __builtin_amdgcn_wmma_f32_16x16x32_bf16(
                false, afrag[kk], false, bfrag, (short)0, acc, false, false);
        }

        // Fused epilogue: tanh(acc + dfb + cov*W_c) * W_v, accumulate score.
        const float dfb_o = dfb[b * H_ + o];
        const float wc_o  = W_c[o];
        const float wv_o  = W_v[o];
        #pragma unroll
        for (int r = 0; r < 8; ++r) {
            float e = tanhf(acc[r] + dfb_o + cov[r] * wc_o);
            sc[r] = fmaf(e, wv_o, sc[r]);
        }
    }

    // Reduce over the 16 feature lanes in each half (butterfly stays in-half).
    #pragma unroll
    for (int m = 1; m < 16; m <<= 1) {
        #pragma unroll
        for (int r = 0; r < 8; ++r)
            sc[r] += __shfl_xor(sc[r], m, 32);
    }
    if ((lane & 15) == 0) {
        #pragma unroll
        for (int r = 0; r < 8; ++r)
            scores[(size_t)b * S_ + tok0 + 8 * half + r] = sc[r];
    }
}

// ---------------------------------------------------------------------------
// Kernel 4: masked softmax over S + coverage update.
// attn = exp(s-max)*mask / sum(exp(s-max)*mask)  (softmax denom cancels)
// ---------------------------------------------------------------------------
__global__ void softmax_kernel(const float* __restrict__ scores,
                               const float* __restrict__ mask,
                               const float* __restrict__ coverage,
                               float* __restrict__ attn,
                               float* __restrict__ cov_out) {
    int b = blockIdx.x;
    int tid = threadIdx.x;                       // 256 threads, 8 elems each
    __shared__ float red[256];

    float v[8];
    float mx = -INFINITY;
    #pragma unroll
    for (int i = 0; i < 8; ++i) {
        v[i] = scores[(size_t)b * S_ + i * 256 + tid];
        mx = fmaxf(mx, v[i]);
    }
    red[tid] = mx; __syncthreads();
    for (int off = 128; off > 0; off >>= 1) {
        if (tid < off) red[tid] = fmaxf(red[tid], red[tid + off]);
        __syncthreads();
    }
    float M = red[0];
    __syncthreads();

    float sum = 0.f;
    #pragma unroll
    for (int i = 0; i < 8; ++i) {
        float e = __expf(v[i] - M) * mask[(size_t)b * S_ + i * 256 + tid];
        v[i] = e;
        sum += e;
    }
    red[tid] = sum; __syncthreads();
    for (int off = 128; off > 0; off >>= 1) {
        if (tid < off) red[tid] += red[tid + off];
        __syncthreads();
    }
    float inv = 1.f / red[0];

    #pragma unroll
    for (int i = 0; i < 8; ++i) {
        size_t idx = (size_t)b * S_ + i * 256 + tid;
        float a = v[i] * inv;
        attn[idx] = a;
        cov_out[idx] = coverage[idx] + a;
    }
}

// ---------------------------------------------------------------------------
// Kernel 5: c_t[b,h] = sum_s attn[b,s] * enc[b,s,h]  (streaming, coalesced)
// ---------------------------------------------------------------------------
__global__ void context_kernel(const float* __restrict__ enc,
                               const float* __restrict__ attn,
                               float* __restrict__ c_t) {
    int b = blockIdx.x >> 1;
    int h = ((blockIdx.x & 1) << 8) + threadIdx.x;   // 256 h per block
    __shared__ float a[S_];
    #pragma unroll
    for (int i = 0; i < 8; ++i)
        a[i * 256 + threadIdx.x] = attn[(size_t)b * S_ + i * 256 + threadIdx.x];
    __syncthreads();

    const float* base = enc + (size_t)b * S_ * H_ + h;
    float acc = 0.f;
    #pragma unroll 8
    for (int s = 0; s < S_; ++s)
        acc = fmaf(a[s], base[(size_t)s * H_], acc);
    c_t[b * H_ + h] = acc;
}

// ---------------------------------------------------------------------------
extern "C" void kernel_launch(void* const* d_in, const int* in_sizes, int n_in,
                              void* d_out, int out_size, void* d_ws, size_t ws_size,
                              hipStream_t stream) {
    const float* enc    = (const float*)d_in[0];   // [B,S,H]
    const float* mask   = (const float*)d_in[1];   // [B,S]
    const float* s_t    = (const float*)d_in[2];   // [B,H]
    const float* cover  = (const float*)d_in[3];   // [B,S]
    const float* W_enc  = (const float*)d_in[4];   // [H,H]
    const float* b_enc  = (const float*)d_in[5];   // [H]
    const float* W_dec  = (const float*)d_in[6];   // [H,H]
    const float* b_dec  = (const float*)d_in[7];   // [H]
    const float* W_c    = (const float*)d_in[8];   // [H,1]
    const float* W_v    = (const float*)d_in[9];   // [1,H]

    // Workspace layout (< 1 MB total)
    unsigned short* Wbf = (unsigned short*)d_ws;                       // 512 KB
    float* dfb    = (float*)((char*)d_ws + (size_t)H_ * H_ * 2);       //  64 KB
    float* scores = (float*)((char*)d_ws + (size_t)H_ * H_ * 2
                                         + (size_t)B_ * H_ * 4);       // 256 KB

    // Output layout: c_t [B,H] | attn_dist [B,S] | coverage_out [B,S]
    float* out     = (float*)d_out;
    float* c_t     = out;
    float* attn    = out + B_ * H_;
    float* cov_out = out + B_ * H_ + B_ * S_;

    wenc_to_bf16_kernel<<<(H_ * H_ / 4) / 256, 256, 0, stream>>>(W_enc, Wbf);
    dec_kernel<<<B_, 256, 0, stream>>>(s_t, W_dec, b_dec, b_enc, dfb);
    scores_kernel<<<B_ * (S_ / 128), 256, 0, stream>>>(enc, Wbf, dfb, cover,
                                                       W_c, W_v, scores);
    softmax_kernel<<<B_, 256, 0, stream>>>(scores, mask, cover, attn, cov_out);
    context_kernel<<<B_ * 2, 256, 0, stream>>>(enc, attn, c_t);
}
